// CPSFFusedCodebook_64965675319784
// MI455X (gfx1250) — compile-verified, matrix operand order checked by reading, not run on hardware
//
#include <hip/hip_runtime.h>
#include <hip/hip_bf16.h>

// ---------------------------------------------------------------------------
// Problem constants (match reference)
// ---------------------------------------------------------------------------
#define BQ 1024   // queries
#define NF 128    // complex ambient dim
#define MC 8192   // codebook entries
#define SD 256    // payload dim
#define KCHUNK 128  // k-elements staged per TDM chunk in out_gemm

#define PI_F        3.14159265358979323846f
#define INV_SQRT_PI 0.5641895835477563f
#define TINY_F      1.17549435e-38f
#define EPS_TOTAL_F 1e-3f

// Tensor Data Mover available? (5-arg on ROCm7.2/clang-22, 6-arg on clang-23)
#if defined(__HIP_DEVICE_COMPILE__) && __has_builtin(__builtin_amdgcn_tensor_load_to_lds)
#define HAVE_TDM 1
#else
#define HAVE_TDM 0
#endif

// ---------------------------------------------------------------------------
// WMMA types & helpers (CDNA5 wave32, 16x16x32 bf16 -> f32)
// ---------------------------------------------------------------------------
typedef __attribute__((ext_vector_type(16))) __bf16         bf16x16;
typedef __attribute__((ext_vector_type(8)))  float          f32x8;
typedef __attribute__((ext_vector_type(8)))  unsigned short u16x8;
typedef __attribute__((ext_vector_type(16))) unsigned short u16x16;
typedef __attribute__((ext_vector_type(4)))  unsigned int   u32x4;
typedef __attribute__((ext_vector_type(8)))  int            i32x8;
typedef __attribute__((ext_vector_type(4)))  int            i32x4;

// 16-bit A/B fragment: lane owns one stored row (A: output-row / B: stored^T
// row).  Per ISA: lanes 0-15 hold K {0..7, 16..23}, lanes 16-31 hold
// K {8..15, 24..31} -> two contiguous 16B loads per lane (global or LDS).
__device__ __forceinline__ bf16x16 load_frag(const __bf16* __restrict__ row_ptr,
                                             int k0, int lane) {
  int kh = k0 + ((lane & 16) ? 8 : 0);
  u16x8 lo = *(const u16x8*)(row_ptr + kh);
  u16x8 hi = *(const u16x8*)(row_ptr + kh + 16);
  u16x16 v;
#pragma unroll
  for (int i = 0; i < 8; ++i) { v[i] = lo[i]; v[i + 8] = hi[i]; }
  return __builtin_bit_cast(bf16x16, v);
}

__device__ __forceinline__ f32x8 wmma_bf16(bf16x16 a, bf16x16 b, f32x8 c) {
  // (neg_a, A, neg_b, B, c_mod, C, reuse_a, reuse_b)
  return __builtin_amdgcn_wmma_f32_16x16x32_bf16(false, a, false, b,
                                                 (short)0, c, false, false);
}

// Generic LDS pointer -> byte offset within LDS (low 32 bits of the aperture
// address).  Also forces the shared array's address to escape into the TDM
// intrinsic so the compiler cannot fold LDS loads of it to undef.
__device__ __forceinline__ unsigned lds_offset_of(const void* p) {
  return (unsigned)(unsigned long long)p;
}

// ---------------------------------------------------------------------------
// TDM: 2D tile (tile_rows x tile_k bf16 elements) from global -> LDS.
// Builds the D# per CDNA5 ISA: group0 = {count=1, lds_addr, global_addr,
// type=2}; group1 = {data_size=1(2B), tensor dims, tile dims, dim0 stride}.
// Issue from ONE wave only; completion via TENSORcnt.
// ---------------------------------------------------------------------------
__device__ __forceinline__ void tdm_load_2d(unsigned lds_off, const void* gptr,
                                            unsigned tile_k, unsigned tile_rows,
                                            unsigned tensor_k, unsigned tensor_rows,
                                            unsigned row_stride_elems) {
#if HAVE_TDM
  unsigned long long ga = (unsigned long long)gptr;
  u32x4 g0;
  g0[0] = 1u;                                     // count=1 user descriptor
  g0[1] = lds_off;                                // lds_addr (bytes)
  g0[2] = (unsigned)(ga & 0xffffffffu);           // global_addr[31:0]
  g0[3] = (unsigned)((ga >> 32) & 0x01ffffffu)    // global_addr[56:32]
        | (2u << 30);                             // type = 2 ("image")
  i32x8 g1;
  g1[0] = (int)(1u << 16);                        // data_size=1 -> 2 bytes
  g1[1] = (int)((tensor_k & 0xffffu) << 16);      // tensor_dim0[15:0]
  g1[2] = (int)(((tensor_k >> 16) & 0xffffu) |    // tensor_dim0[31:16]
                ((tensor_rows & 0xffffu) << 16)); // tensor_dim1[15:0]
  g1[3] = (int)(((tensor_rows >> 16) & 0xffffu) | // tensor_dim1[31:16]
                ((tile_k & 0xffffu) << 16));      // tile_dim0
  g1[4] = (int)(tile_rows & 0xffffu);             // tile_dim1 (tile_dim2=0)
  g1[5] = (int)row_stride_elems;                  // tensor_dim0_stride[31:0]
  g1[6] = 0;                                      // stride[47:32], dim1_stride
  g1[7] = 0;
  i32x4 z4 = {0, 0, 0, 0};
#if defined(__clang_major__) && (__clang_major__ >= 23)
  i32x8 z8 = {0, 0, 0, 0, 0, 0, 0, 0};
  __builtin_amdgcn_tensor_load_to_lds(g0, g1, z4, z4, z8, 0);
#else
  __builtin_amdgcn_tensor_load_to_lds(g0, g1, z4, z4, 0);
#endif
#else
  (void)lds_off; (void)gptr; (void)tile_k; (void)tile_rows;
  (void)tensor_k; (void)tensor_rows; (void)row_stride_elems;
#endif
}

__device__ __forceinline__ void tdm_wait0() {
#if HAVE_TDM
  __builtin_amdgcn_s_wait_tensorcnt(0);
#endif
}

// block of 128 threads (4 waves) -> scalar broadcast sum
__device__ __forceinline__ float block_reduce_128(float v, float* sm) {
#pragma unroll
  for (int m = 16; m >= 1; m >>= 1) v += __shfl_xor(v, m, 32);
  int wid = threadIdx.x >> 5;
  if ((threadIdx.x & 31) == 0) sm[wid] = v;
  __syncthreads();
  float r = sm[0] + sm[1] + sm[2] + sm[3];
  __syncthreads();
  return r;
}

// ---------------------------------------------------------------------------
// Kernel 1: query-side prep.  grid = B, block = 128 (one thread per feature)
// ---------------------------------------------------------------------------
__global__ void __launch_bounds__(128) prep_queries(
    const float* __restrict__ zre, const float* __restrict__ zim,
    const float* __restrict__ dre, const float* __restrict__ dim_,
    __bf16* __restrict__ zr_bf, __bf16* __restrict__ zi_bf,
    __bf16* __restrict__ ur_bf, __bf16* __restrict__ ui_bf,
    float* __restrict__ z2, float* __restrict__ den) {
  __shared__ float sm[4];
  int b = blockIdx.x, t = threadIdx.x;
  int idx = b * NF + t;
  float zr = zre[idx], zi = zim[idx], dr = dre[idx], di = dim_[idx];
  float dn2 = block_reduce_128(dr * dr + di * di, sm);
  float nrm = sqrtf(dn2);
  float inv = (nrm == 0.f) ? 1.f : 1.f / nrm;
  float zn2 = block_reduce_128(zr * zr + zi * zi, sm);
  zr_bf[idx] = (__bf16)zr;
  zi_bf[idx] = (__bf16)zi;
  ur_bf[idx] = (__bf16)(dr * inv);
  ui_bf[idx] = (__bf16)(di * inv);
  if (t == 0) { z2[b] = zn2; den[b] = EPS_TOTAL_F; }
}

// ---------------------------------------------------------------------------
// Kernel 2: codebook-side prep.  grid = M, block = 128
// params SoA: [0]=cr [1]=ci [2]=|zj|^2 [3]=alpha [4]=pi/sp [5]=pi/sq
//             [6]=scale [7]=scale/sqrt(pi)
// ---------------------------------------------------------------------------
__global__ void __launch_bounds__(128) prep_codebook(
    const float* __restrict__ zjre, const float* __restrict__ zjim,
    const float* __restrict__ djre, const float* __restrict__ djim,
    const float* __restrict__ alpha, const float* __restrict__ sigp,
    const float* __restrict__ sigq,
    __bf16* __restrict__ djr_bf, __bf16* __restrict__ dji_bf,
    __bf16* __restrict__ zjr_bf, __bf16* __restrict__ zji_bf,
    float* __restrict__ params) {
  __shared__ float sm[4];
  int m = blockIdx.x, t = threadIdx.x;
  int idx = m * NF + t;
  float zr = zjre[idx], zi = zjim[idx], dr = djre[idx], di = djim[idx];
  float dn2 = block_reduce_128(dr * dr + di * di, sm);
  float nrm = sqrtf(dn2);
  float inv = (nrm == 0.f) ? 1.f : 1.f / nrm;
  dr *= inv; di *= inv;
  float cr  = block_reduce_128(dr * zr + di * zi, sm);
  float ci  = block_reduce_128(dr * zi - di * zr, sm);
  float zj2 = block_reduce_128(zr * zr + zi * zi, sm);
  djr_bf[idx] = (__bf16)dr;
  dji_bf[idx] = (__bf16)di;
  zjr_bf[idx] = (__bf16)zr;
  zji_bf[idx] = (__bf16)zi;
  if (t == 0) {
    float a  = fmaxf(alpha[m], TINY_F);
    float sp = fmaxf(sigp[m], TINY_F);
    float sq = fmaxf(sigq[m], TINY_F);
    float scale = sqrtf(sp / PI_F);
    params[0 * MC + m] = cr;
    params[1 * MC + m] = ci;
    params[2 * MC + m] = zj2;
    params[3 * MC + m] = a;
    params[4 * MC + m] = PI_F / sp;
    params[5 * MC + m] = PI_F / sq;
    params[6 * MC + m] = scale;
    params[7 * MC + m] = scale * INV_SQRT_PI;
  }
}

// ---------------------------------------------------------------------------
// Kernel 3: T[M,S] -> Tt[S,M] bf16 (makes stage-3 B-fragments row-contiguous)
// ---------------------------------------------------------------------------
__global__ void __launch_bounds__(256) prep_T(
    const float* __restrict__ Tre, const float* __restrict__ Tim,
    __bf16* __restrict__ Ttr, __bf16* __restrict__ Tti) {
  int idx = blockIdx.x * 256 + threadIdx.x;  // over M*S
  int m = idx / SD, s = idx % SD;
  Ttr[(size_t)s * MC + m] = (__bf16)Tre[idx];
  Tti[(size_t)s * MC + m] = (__bf16)Tim[idx];
}

// ---------------------------------------------------------------------------
// Kernel 4: fused mixture weights W[b,m] = alpha*align*rho, plus den[b].
// Block = 8 waves (2 b-bands x 4 m-tiles of 16x16); 10 f32 accumulators/wave,
// 40 v_wmma per wave over the N=128 contraction.  The four codebook operand
// tiles (64 rows x 128 k bf16 = 16KB each) are TDM-staged into LDS once per
// block and shared by both b-bands (halves B-side L2 traffic).
// ---------------------------------------------------------------------------
__global__ void __launch_bounds__(256) fuse_weights(
    const __bf16* __restrict__ zr_bf, const __bf16* __restrict__ zi_bf,
    const __bf16* __restrict__ ur_bf, const __bf16* __restrict__ ui_bf,
    const __bf16* __restrict__ djr_bf, const __bf16* __restrict__ dji_bf,
    const __bf16* __restrict__ zjr_bf, const __bf16* __restrict__ zji_bf,
    const float* __restrict__ params, const float* __restrict__ z2,
    float* __restrict__ den, __bf16* __restrict__ Wbf) {
#if HAVE_TDM
  __shared__ __align__(32) __bf16 sCB[4][64 * NF];  // 64KB: djr,dji,zjr,zji
#endif
  int lane = threadIdx.x & 31;
  int wave = threadIdx.x >> 5;
  int wb = wave >> 2, wm = wave & 3;
  int b0 = blockIdx.y * 32 + wb * 16;
  int m0 = blockIdx.x * 64 + wm * 16;

  const __bf16* pzr = zr_bf + (size_t)(b0 + (lane & 15)) * NF;
  const __bf16* pzi = zi_bf + (size_t)(b0 + (lane & 15)) * NF;
  const __bf16* pur = ur_bf + (size_t)(b0 + (lane & 15)) * NF;
  const __bf16* pui = ui_bf + (size_t)(b0 + (lane & 15)) * NF;

#if HAVE_TDM
  // Opaque, never-taken store: blockDim.x is a runtime value, so the compiler
  // must treat sCB as written and cannot fold its loads to undef (the real
  // writes come from the TDM, which it cannot see).
  if (blockDim.x > 1024) sCB[0][threadIdx.x] = (__bf16)0.0f;
  unsigned lds_base = lds_offset_of(&sCB[0][0]);
  if (wave == 0) {
    size_t mrow = (size_t)blockIdx.x * 64 * NF;
    const unsigned tb = 64u * NF * sizeof(__bf16);  // 16KB per tile
    tdm_load_2d(lds_base + 0 * tb, djr_bf + mrow, NF, 64, NF, MC, NF);
    tdm_load_2d(lds_base + 1 * tb, dji_bf + mrow, NF, 64, NF, MC, NF);
    tdm_load_2d(lds_base + 2 * tb, zjr_bf + mrow, NF, 64, NF, MC, NF);
    tdm_load_2d(lds_base + 3 * tb, zji_bf + mrow, NF, 64, NF, MC, NF);
    tdm_wait0();
  }
  __syncthreads();
  int brow = (wm * 16 + (lane & 15)) * NF;  // this lane's codebook row in LDS
  const __bf16* pdr = &sCB[0][brow];
  const __bf16* pdi = &sCB[1][brow];
  const __bf16* pqr = &sCB[2][brow];
  const __bf16* pqi = &sCB[3][brow];
#else
  const __bf16* pdr = djr_bf + (size_t)(m0 + (lane & 15)) * NF;
  const __bf16* pdi = dji_bf + (size_t)(m0 + (lane & 15)) * NF;
  const __bf16* pqr = zjr_bf + (size_t)(m0 + (lane & 15)) * NF;
  const __bf16* pqi = zji_bf + (size_t)(m0 + (lane & 15)) * NF;
#endif

  f32x8 P1{}, P2{}, P3{}, P4{}, Q1{}, Q2{}, Q3{}, Q4{}, R1{}, R2{};
#pragma unroll
  for (int k0 = 0; k0 < NF; k0 += 32) {
    bf16x16 azr = load_frag(pzr, k0, lane);
    bf16x16 azi = load_frag(pzi, k0, lane);
    bf16x16 aur = load_frag(pur, k0, lane);
    bf16x16 aui = load_frag(pui, k0, lane);
    bf16x16 bdr = load_frag(pdr, k0, lane);
    bf16x16 bdi = load_frag(pdi, k0, lane);
    bf16x16 bqr = load_frag(pqr, k0, lane);
    bf16x16 bqi = load_frag(pqi, k0, lane);
    P1 = wmma_bf16(azr, bdr, P1);   // z_re @ djr^T
    P2 = wmma_bf16(azi, bdi, P2);   // z_im @ dji^T
    P3 = wmma_bf16(azi, bdr, P3);   // z_im @ djr^T
    P4 = wmma_bf16(azr, bdi, P4);   // z_re @ dji^T
    Q1 = wmma_bf16(aur, bdr, Q1);   // dur @ djr^T
    Q2 = wmma_bf16(aui, bdi, Q2);   // dui @ dji^T
    Q3 = wmma_bf16(aui, bdr, Q3);   // dui @ djr^T
    Q4 = wmma_bf16(aur, bdi, Q4);   // dur @ dji^T
    R1 = wmma_bf16(azr, bqr, R1);   // z_re @ zj_re^T
    R2 = wmma_bf16(azi, bqi, R2);   // z_im @ zj_im^T
  }

  // C/D fragment map: col = lane&15, row = v + 8*(lane>>4)
  int m   = m0 + (lane & 15);
  float cr  = params[0 * MC + m];
  float ci_ = params[1 * MC + m];
  float zj2 = params[2 * MC + m];
  float am  = params[3 * MC + m];
  float isp = params[4 * MC + m];
  float isq = params[5 * MC + m];
  float scl = params[6 * MC + m];
  float sw  = params[7 * MC + m];
  int rbase = b0 + ((lane >> 4) << 3);

  // Gauss-Hermite K=8, +/- node pairs (constexpr -> inline literals)
  constexpr float ght[4] = {0.3811869902073221f, 1.1571937124467803f,
                            1.9816567566958429f, 2.9306374202572440f};
  constexpr float ghw[4] = {0.6611470125582412f, 0.2078023258148919f,
                            0.0170779830074135f, 1.9960407221136762e-4f};

#pragma unroll
  for (int v = 0; v < 8; ++v) {
    int b = rbase + v;
    float pr = P1[v] + P2[v] - cr;
    float pi = P3[v] - P4[v] - ci_;
    float dz2 = z2[b] + zj2 - 2.f * (R1[v] + R2[v]);
    float perp2 = fmaxf(dz2 - pr * pr - pi * pi, 0.f);
    float alr = Q1[v] + Q2[v];
    float ali = Q3[v] - Q4[v];
    float align2 = alr * alr + ali * ali;
    float base = pi * pi * isp + perp2 * isq;
    float s = 0.f;
#pragma unroll
    for (int j = 0; j < 4; ++j) {   // 8 exps total
      float t  = ght[j] * scl;
      float d0 = pr - t, d1 = pr + t;
      s += ghw[j] * (__expf(-(base + d0 * d0 * isp)) +
                     __expf(-(base + d1 * d1 * isp)));
    }
    float w = am * align2 * s * sw;
    Wbf[(size_t)b * MC + m] = (__bf16)w;
    // lanes 0-15 (and 16-31) share the same b for a given v: butterfly-sum
    float part = w;
    part += __shfl_xor(part, 1, 32);
    part += __shfl_xor(part, 2, 32);
    part += __shfl_xor(part, 4, 32);
    part += __shfl_xor(part, 8, 32);
    if ((lane & 15) == 0) atomicAdd(&den[b], part);
  }
}

// ---------------------------------------------------------------------------
// Kernel 5: out = (W @ T) / den.  K = M = 8192.  Block tile = 32b x 128s,
// 8 waves each owning one 16-col s-tile and BOTH 16-row b-tiles (4
// accumulators, 4 wmma per k-step on shared T fragments -> 1 load per wmma).
// The 32-row W tile is shared by all waves and TDM-staged into LDS in
// double-buffered 32 x KCHUNK chunks (issue c+1, compute c, s_wait_tensorcnt,
// barrier); the streaming T operands stay on the global path with prefetch.
// ---------------------------------------------------------------------------
__global__ void __launch_bounds__(256) out_gemm(
    const __bf16* __restrict__ Wbf, const __bf16* __restrict__ Ttr,
    const __bf16* __restrict__ Tti, const float* __restrict__ den,
    float* __restrict__ out) {
#if HAVE_TDM
  __shared__ __align__(32) __bf16 sW[2][32 * KCHUNK];  // 2 x 8KB
#endif
  int lane = threadIdx.x & 31;
  int wave = threadIdx.x >> 5;
  int s0 = blockIdx.x * 128 + wave * 16;
  int bblk = blockIdx.y * 32;

  const __bf16* pw0 = Wbf + (size_t)(bblk + (lane & 15)) * MC;       // fallback
  const __bf16* pw1 = Wbf + (size_t)(bblk + 16 + (lane & 15)) * MC;  // fallback
  const __bf16* pr = Ttr + (size_t)(s0 + (lane & 15)) * MC;
  const __bf16* pi = Tti + (size_t)(s0 + (lane & 15)) * MC;
  (void)pw0; (void)pw1;

  const int NCH = MC / KCHUNK;

#if HAVE_TDM
  if (blockDim.x > 1024) sW[0][threadIdx.x] = (__bf16)0.0f;  // opaque store
  unsigned lds_base = lds_offset_of(&sW[0][0]);
  const unsigned bufb = 32u * KCHUNK * sizeof(__bf16);  // 8KB per buffer
  if (wave == 0) {
    tdm_load_2d(lds_base, Wbf + (size_t)bblk * MC, KCHUNK, 32, MC, BQ, MC);
    tdm_wait0();
  }
  __syncthreads();
#endif

  f32x8 accR0{}, accI0{}, accR1{}, accI1{};
  for (int c = 0; c < NCH; ++c) {
    int kbase = c * KCHUNK;
#if HAVE_TDM
    int cur = c & 1;
    if (wave == 0 && (c + 1) < NCH) {
      tdm_load_2d(lds_base + bufb * (unsigned)(cur ^ 1),
                  Wbf + (size_t)bblk * MC + (size_t)(c + 1) * KCHUNK,
                  KCHUNK, 32, MC, BQ, MC);
    }
    const __bf16* arow0 = &sW[cur][(lane & 15) * KCHUNK];
    const __bf16* arow1 = &sW[cur][(16 + (lane & 15)) * KCHUNK];
#endif
    if (kbase + KCHUNK < MC) {  // stream-ahead hint for the T operands
      __builtin_prefetch(pr + kbase + KCHUNK, 0, 1);
      __builtin_prefetch(pi + kbase + KCHUNK, 0, 1);
    }
#pragma unroll
    for (int kk = 0; kk < KCHUNK; kk += 32) {
      int k0 = kbase + kk;
#if HAVE_TDM
      bf16x16 a0 = load_frag(arow0, kk, lane);
      bf16x16 a1 = load_frag(arow1, kk, lane);
#else
      bf16x16 a0 = load_frag(pw0, k0, lane);
      bf16x16 a1 = load_frag(pw1, k0, lane);
#endif
      bf16x16 br = load_frag(pr, k0, lane);
      bf16x16 bi = load_frag(pi, k0, lane);
      accR0 = wmma_bf16(a0, br, accR0);
      accI0 = wmma_bf16(a0, bi, accI0);
      accR1 = wmma_bf16(a1, br, accR1);
      accI1 = wmma_bf16(a1, bi, accI1);
    }
#if HAVE_TDM
    if (wave == 0) tdm_wait0();   // chunk c+1 resident before anyone reads it
    __syncthreads();              // and everyone done reading chunk c
#endif
  }

  int s = s0 + (lane & 15);
#pragma unroll
  for (int t = 0; t < 2; ++t) {
    const f32x8& aR = t ? accR1 : accR0;
    const f32x8& aI = t ? accI1 : accI0;
    int rbase = bblk + t * 16 + ((lane >> 4) << 3);
#pragma unroll
    for (int v = 0; v < 8; ++v) {
      int b = rbase + v;
      float r = 1.f / den[b];
      float2 o = make_float2(aR[v] * r, aI[v] * r);   // [B,S,2] packed
      *reinterpret_cast<float2*>(out + ((size_t)b * SD + s) * 2) = o;
    }
  }
}

// ---------------------------------------------------------------------------
// Host launcher
// ---------------------------------------------------------------------------
extern "C" void kernel_launch(void* const* d_in, const int* in_sizes, int n_in,
                              void* d_out, int out_size, void* d_ws, size_t ws_size,
                              hipStream_t stream) {
  (void)in_sizes; (void)n_in; (void)out_size; (void)ws_size;
  const float* z_re  = (const float*)d_in[0];
  const float* z_im  = (const float*)d_in[1];
  const float* d_re  = (const float*)d_in[2];
  const float* d_im  = (const float*)d_in[3];
  const float* zj_re = (const float*)d_in[4];
  const float* zj_im = (const float*)d_in[5];
  const float* dj_re = (const float*)d_in[6];
  const float* dj_im = (const float*)d_in[7];
  const float* T_re  = (const float*)d_in[8];
  const float* T_im  = (const float*)d_in[9];
  const float* alpha = (const float*)d_in[10];
  const float* sigp  = (const float*)d_in[11];
  const float* sigq  = (const float*)d_in[12];
  float* out = (float*)d_out;

  char* ws = (char*)d_ws;
  size_t off = 0;
  auto alloc = [&](size_t bytes) -> void* {
    off = (off + 255) & ~(size_t)255;
    void* p = ws + off;
    off += bytes;
    return p;
  };

  __bf16* zr_bf  = (__bf16*)alloc((size_t)BQ * NF * 2);
  __bf16* zi_bf  = (__bf16*)alloc((size_t)BQ * NF * 2);
  __bf16* ur_bf  = (__bf16*)alloc((size_t)BQ * NF * 2);
  __bf16* ui_bf  = (__bf16*)alloc((size_t)BQ * NF * 2);
  __bf16* djr_bf = (__bf16*)alloc((size_t)MC * NF * 2);
  __bf16* dji_bf = (__bf16*)alloc((size_t)MC * NF * 2);
  __bf16* zjr_bf = (__bf16*)alloc((size_t)MC * NF * 2);
  __bf16* zji_bf = (__bf16*)alloc((size_t)MC * NF * 2);
  __bf16* Ttr    = (__bf16*)alloc((size_t)SD * MC * 2);
  __bf16* Tti    = (__bf16*)alloc((size_t)SD * MC * 2);
  __bf16* Wbf    = (__bf16*)alloc((size_t)BQ * MC * 2);
  float*  params = (float*)alloc((size_t)8 * MC * 4);
  float*  z2     = (float*)alloc((size_t)BQ * 4);
  float*  den    = (float*)alloc((size_t)BQ * 4);

  prep_queries<<<BQ, 128, 0, stream>>>(z_re, z_im, d_re, d_im,
                                       zr_bf, zi_bf, ur_bf, ui_bf, z2, den);
  prep_codebook<<<MC, 128, 0, stream>>>(zj_re, zj_im, dj_re, dj_im,
                                        alpha, sigp, sigq,
                                        djr_bf, dji_bf, zjr_bf, zji_bf, params);
  prep_T<<<(MC * SD) / 256, 256, 0, stream>>>(T_re, T_im, Ttr, Tti);
  fuse_weights<<<dim3(MC / 64, BQ / 32), 256, 0, stream>>>(
      zr_bf, zi_bf, ur_bf, ui_bf, djr_bf, dji_bf, zjr_bf, zji_bf,
      params, z2, den, Wbf);
  out_gemm<<<dim3(SD / 128, BQ / 32), 256, 0, stream>>>(Wbf, Ttr, Tti, den, out);
}
